// CIFARClassifier_6279242187503
// MI455X (gfx1250) — matrix operand order repaired and checked by reading, not compiled, not run on hardware
//
#include <hip/hip_runtime.h>
#include <math.h>

typedef _Float16 h16;
typedef __attribute__((ext_vector_type(2)))  _Float16 h16x2;
typedef __attribute__((ext_vector_type(8)))  _Float16 v8h;
typedef __attribute__((ext_vector_type(16))) _Float16 v16h;
typedef __attribute__((ext_vector_type(8)))  float    v8f;

#define WAVES_PER_BLK 4
#define THREADS (WAVES_PER_BLK * 32)

// Halo-padded plane sizes: (Wd+2)^2 per channel, 3 channels per buffer.
#define PL0 (34*34)            // 1156
#define PL1 (18*18)            // 324
#define PL2 (10*10)            // 100
// Buffer allocation strides padded to multiples of 64 f16 so elementwise
// sweeps have lane-uniform trip counts (no divergent tails).
#define SA 3520                // >= 3*PL0 = 3468
#define SB 1024                // >= 3*PL1 = 972
#define SC 320                 // >= 3*PL2 = 300
#define WAVE_E 16128           // >= 3*SA+4*SB+4*SC = 15936 ; = 63*8*32 (uniform zero-init)

static __device__ __forceinline__ void lds_fence() {
  asm volatile("s_wait_dscnt 0x0" ::: "memory");
}

// 3x3 SAME conv (Cin=3 -> Cout=3) + folded BN + ReLU on halo-padded f16 LDS.
// Implicit GEMM per tile: M=16 pixels, N=16 (3 live), K=32 (27 live; k>=27
// multiplied by B-side zeros, so A may carry arbitrary finite data there).
template<int Wd>
static __device__ void conv_blk(const h16* src, h16* dst,
                                const float* __restrict__ wl,   // [cout][27] f32
                                const float* __restrict__ gm, const float* __restrict__ bt,
                                const float* __restrict__ mn, const float* __restrict__ vr,
                                int layer, int lane)
{
  const int PW = Wd + 2, Pp = PW * PW;
  const int P = Wd * Wd, T = P / 16;
  const int n  = lane & 15;
  const int lo = (lane < 16);

  // Fold eval-mode BN into weight scale + accumulator bias
  float sc = 0.f, bs = 0.f;
  if (n < 3) {
    float inv = gm[layer*3 + n] * rsqrtf(vr[layer*3 + n] + 1e-5f);
    sc = inv;
    bs = bt[layer*3 + n] - mn[layer*3 + n] * inv;
  }

  // B fragment (32x16 f16): lanes 0-15 hold K=0..15 (N=lane&15), lanes 16-31 K=16..31
  v16h Bm;
  {
    const int kb = lo ? 0 : 16;
    #pragma unroll
    for (int v = 0; v < 8; ++v) {
      int k0 = kb + 2*v, k1 = k0 + 1;
      float f0 = (n < 3 && k0 < 27) ? wl[n*27 + k0] * sc : 0.f;
      float f1 = (n < 3 && k1 < 27) ? wl[n*27 + k1] * sc : 0.f;
      Bm[2*v]   = (h16)f0;
      Bm[2*v+1] = (h16)f1;
    }
  }

  // Loop-invariant per-lane gather offsets for the A fragment.
  // A 16x32 layout: lanes 0-15 carry K 0-15, lanes 16-31 K 16-31.
  // k>=27 clamps to k-27 (finite dummy; B column is zero there).
  int off[16];
  #pragma unroll
  for (int v = 0; v < 8; ++v) {
    int k0 = (lo ? 0 : 16) + 2*v;
    #pragma unroll
    for (int j = 0; j < 2; ++j) {
      int k  = k0 + j;
      int kc = (k < 27) ? k : (k - 27);
      int c  = kc / 9, r = kc - 9*c;
      off[2*v + j] = c*Pp + (r/3)*PW + (r%3) - PW - 1;   // dy*PW + dx
    }
  }

  const int mrow = lane & 15;
  #pragma unroll 2
  for (int t = 0; t < T; ++t) {
    const int p = t*16 + mrow;
    const int y = p / Wd, x = p % Wd;
    const int base = (y + 1)*PW + (x + 1);
    v16h Am;
    #pragma unroll
    for (int e = 0; e < 16; ++e)
      Am[e] = src[base + off[e]];              // unconditional ds_load_u16

    v8f Cm;
    #pragma unroll
    for (int r = 0; r < 8; ++r) Cm[r] = bs;
    Cm = __builtin_amdgcn_wmma_f32_16x16x32_f16(false, Am, false, Bm,
                                                (short)0, Cm, false, false);

    if (n < 3) {                               // 6 live lanes, 8 row-contiguous pixels
      const int p0  = t*16 + (lo ? 0 : 8);
      const int row = p0 / Wd, col = p0 % Wd;
      h16* d = dst + n*Pp + (row + 1)*PW + (col + 1);
      #pragma unroll
      for (int r = 0; r < 8; ++r) {
        float v2 = Cm[r];
        v2 = v2 > 0.f ? v2 : 0.f;
        d[r] = (h16)v2;
      }
    }
  }
  lds_fence();
}

// Packed-f16 elementwise sweeps over padded-stride buffers.
// N is a multiple of 64 f16 -> N/2 pairs is a multiple of 32: uniform loops.
template<int N>
static __device__ void add_inplace(h16* dst, const h16* a, int lane) {
  h16x2* d2 = (h16x2*)dst; const h16x2* a2 = (const h16x2*)a;
  for (int j = 0; j < N/64; ++j) { int i = lane + 32*j; d2[i] = d2[i] + a2[i]; }
  lds_fence();
}
template<int N>
static __device__ void add_to(h16* dst, const h16* a, const h16* b, int lane) {
  h16x2* d2 = (h16x2*)dst; const h16x2* a2 = (const h16x2*)a; const h16x2* b2 = (const h16x2*)b;
  for (int j = 0; j < N/64; ++j) { int i = lane + 32*j; d2[i] = a2[i] + b2[i]; }
  lds_fence();
}
template<int N>
static __device__ void add3_to(h16* dst, const h16* a, const h16* b, const h16* c, int lane) {
  h16x2* d2 = (h16x2*)dst; const h16x2* a2 = (const h16x2*)a;
  const h16x2* b2 = (const h16x2*)b; const h16x2* c2 = (const h16x2*)c;
  for (int j = 0; j < N/64; ++j) { int i = lane + 32*j; d2[i] = a2[i] + b2[i] + c2[i]; }
  lds_fence();
}

template<int Wd>   // input (unpadded) width; both src and dst halo-padded
static __device__ void maxpool2(const h16* src, h16* dst, int lane) {
  const int PW = Wd + 2, Pp = PW * PW;
  const int Wo = Wd / 2, PWo = Wo + 2, Ppo = PWo * PWo, Po = Wo * Wo;
  const int total = 3 * Po;
  #pragma unroll 1
  for (int j = 0; j < (total + 31) / 32; ++j) {
    int i = lane + 32*j;
    if ((total % 32 == 0) || (i < total)) {    // compile-time true for Wd=32,16
      int c = i / Po, rr = i - c*Po, y = rr / Wo, x = rr % Wo;
      const h16* s = src + c*Pp + (2*y + 1)*PW + (2*x + 1);
      float a = (float)s[0], b = (float)s[1], d = (float)s[PW], e = (float)s[PW + 1];
      dst[c*Ppo + (y + 1)*PWo + (x + 1)] = (h16)fmaxf(fmaxf(a, b), fmaxf(d, e));
    }
  }
  lds_fence();
}

__global__ void __launch_bounds__(THREADS)
cifar_fused_kernel(const float* __restrict__ x,
                   const float* __restrict__ ws,
                   const float* __restrict__ w9,
                   const float* __restrict__ gm,
                   const float* __restrict__ bt,
                   const float* __restrict__ mn,
                   const float* __restrict__ vr,
                   float* __restrict__ out,
                   int nimg)
{
  extern __shared__ h16 smem[];
  const int wave = threadIdx.x >> 5;
  const int lane = threadIdx.x & 31;
  const int img  = blockIdx.x * WAVES_PER_BLK + wave;
  if (img >= nimg) return;   // wave-uniform

  h16* m = smem + wave * WAVE_E;   // private LDS slice per wave (one image)

  // Zero the whole slice once (halo borders + slack); uniform b128 loop.
  {
    v8h z = {};
    v8h* mz = (v8h*)m;
    for (int j = 0; j < WAVE_E/(8*32); ++j) mz[lane + 32*j] = z;
    lds_fence();
  }

  h16 *A0 = m,       *A1 = A0 + SA, *A2 = A1 + SA;
  h16 *B0 = A2 + SA, *B1 = B0 + SB, *B2 = B1 + SB, *B3 = B2 + SB;
  h16 *C0 = B3 + SB, *C1 = C0 + SC, *C2 = C1 + SC, *C3 = C2 + SC;

  // Single HBM read of the image: f32 -> f16 into padded LDS interior
  {
    const float4* g = (const float4*)(x + (size_t)img * 3072);
    for (int j = 0; j < 768/32; ++j) {
      int i = lane + 32*j;
      float4 f = g[i];
      int e0 = 4*i;
      int c = e0 >> 10, rr = e0 & 1023;
      int y = rr >> 5, xcol = rr & 31;
      h16* d = A0 + c*PL0 + (y + 1)*34 + (xcol + 1);
      d[0] = (h16)f.x; d[1] = (h16)f.y; d[2] = (h16)f.z; d[3] = (h16)f.w;
    }
    lds_fence();
  }

  // Stage 1: 32x32
  conv_blk<32>(A0, A1, ws + 0*81, gm, bt, mn, vr, 0, lane);   // x2
  add_inplace<SA>(A1, A0, lane);                              // x1+x2
  conv_blk<32>(A1, A2, ws + 1*81, gm, bt, mn, vr, 1, lane);   // x3
  add_inplace<SA>(A2, A1, lane);                              // x1+x2+x3
  maxpool2<32>(A2, B0, lane);                                 // x4

  // Stage 2: 16x16
  conv_blk<16>(B0, B1, ws + 2*81, gm, bt, mn, vr, 2, lane);   // x5
  add_to<SB>(B2, B0, B1, lane);                               // x4+x5
  conv_blk<16>(B2, B3, ws + 3*81, gm, bt, mn, vr, 3, lane);   // x6
  add_inplace<SB>(B2, B3, lane);                              // x4+x5+x6
  conv_blk<16>(B2, B0, ws + 4*81, gm, bt, mn, vr, 4, lane);   // x7 (x4 dead)
  add3_to<SB>(B2, B1, B3, B0, lane);                          // x5+x6+x7
  maxpool2<16>(B2, C0, lane);                                 // x8

  // Stage 3: 8x8
  conv_blk<8>(C0, C1, ws + 5*81, gm, bt, mn, vr, 5, lane);    // x9
  add_to<SC>(C2, C0, C1, lane);                               // x8+x9
  conv_blk<8>(C2, C3, ws + 6*81, gm, bt, mn, vr, 6, lane);    // x10
  add_inplace<SC>(C2, C3, lane);                              // x8+x9+x10
  conv_blk<8>(C2, C0, ws + 7*81, gm, bt, mn, vr, 7, lane);    // x11

  // Head: GAP(8x8) -> 1x1 conv (center tap only) -> log_softmax
  float mean[3];
  #pragma unroll
  for (int c = 0; c < 3; ++c) {
    float s = 0.f;
    #pragma unroll
    for (int j = 0; j < 2; ++j) {
      int i = lane + 32*j;
      int y = i >> 3, xcol = i & 7;
      s += (float)C0[c*PL2 + (y + 1)*10 + (xcol + 1)];
    }
    #pragma unroll
    for (int d = 16; d >= 1; d >>= 1) s += __shfl_xor(s, d, 32);
    mean[c] = s * (1.f / 64.f);
  }
  float lg = -__builtin_inff();
  if (lane < 10) {
    float acc = 0.f;
    #pragma unroll
    for (int c = 0; c < 3; ++c) acc += mean[c] * w9[lane*27 + c*9 + 4];
    lg = acc;
  }
  float mx = lg;
  #pragma unroll
  for (int d = 8; d >= 1; d >>= 1) mx = fmaxf(mx, __shfl_xor(mx, d, 16));
  float ex = (lane < 10) ? expf(lg - mx) : 0.f;
  float se = ex;
  #pragma unroll
  for (int d = 8; d >= 1; d >>= 1) se += __shfl_xor(se, d, 16);
  if (lane < 10) out[(size_t)img*10 + lane] = lg - mx - logf(se);
}

extern "C" void kernel_launch(void* const* d_in, const int* in_sizes, int n_in,
                              void* d_out, int out_size, void* d_ws, size_t ws_size,
                              hipStream_t stream) {
  const float* x  = (const float*)d_in[0];   // [B,3,32,32]
  const float* ws = (const float*)d_in[1];   // [8,3,3,3,3]
  const float* w9 = (const float*)d_in[2];   // [10,3,3,3]
  const float* gm = (const float*)d_in[3];
  const float* bt = (const float*)d_in[4];
  const float* mn = (const float*)d_in[5];
  const float* vr = (const float*)d_in[6];
  float* out = (float*)d_out;

  const int nimg   = in_sizes[0] / 3072;                           // 4096
  const int blocks = (nimg + WAVES_PER_BLK - 1) / WAVES_PER_BLK;   // 1024
  const size_t lds = (size_t)WAVES_PER_BLK * WAVE_E * sizeof(h16); // 126 KB

  cifar_fused_kernel<<<blocks, THREADS, lds, stream>>>(x, ws, w9, gm, bt, mn, vr, out, nimg);
}
